// LmulLinear_32435593019491
// MI455X (gfx1250) — matrix operand order repaired and checked by reading, not compiled, not run on hardware
//
#include <hip/hip_runtime.h>
#include <hip/hip_bf16.h>

typedef _Float16 half16 __attribute__((ext_vector_type(16)));
typedef _Float16 half8  __attribute__((ext_vector_type(8)));
typedef float    float8 __attribute__((ext_vector_type(8)));

// Problem dims (fixed by the reference)
constexpr int TN = 256;   // tokens
constexpr int TK = 512;   // in_features
constexpr int TM = 512;   // out_features
constexpr int KT = 2 * TK; // doubled-K GEMM (1024)
constexpr float LMUL_OFFSET = 0.0625f; // 2^-4 (mantissa_bits=10 -> l_m=4)

// ---------------------------------------------------------------------------
// Prep: A' = [Xq | Ex]  (TN x KT, f16 row-major)
//   Xq = round(mant*1024)/1024 * 2^e ; Ex = 2^e ; both forced to 0 when x==0
// ---------------------------------------------------------------------------
__global__ void lmul_prep_x(const float* __restrict__ x, _Float16* __restrict__ Ap) {
    int i = blockIdx.x * blockDim.x + threadIdx.x;
    if (i >= TN * TK) return;
    int n = i / TK, k = i % TK;
    float v = x[i];
    float xq = 0.0f, ex = 0.0f;
    if (v != 0.0f) {
        int e;
        float m  = frexpf(v, &e);                      // m in [0.5,1), v = m*2^e
        float mq = rintf(m * 1024.0f) * (1.0f / 1024.0f); // RNE, matches jnp.round
        xq = ldexpf(mq, e);
        ex = ldexpf(1.0f, e);
    }
    Ap[(size_t)n * KT + k]      = (_Float16)xq;
    Ap[(size_t)n * KT + TK + k] = (_Float16)ex;
}

// ---------------------------------------------------------------------------
// Prep: B' = [Ew | Wq + c*Ew]  (TM x KT, f16 row-major)
//   Wq + c*Ew = (mq + 0.0625)*2^e  — exactly representable in f16
// ---------------------------------------------------------------------------
__global__ void lmul_prep_w(const float* __restrict__ w, _Float16* __restrict__ Bp) {
    int i = blockIdx.x * blockDim.x + threadIdx.x;
    if (i >= TM * TK) return;
    int m = i / TK, k = i % TK;
    float v = w[i];
    float ew = 0.0f, b2 = 0.0f;
    if (v != 0.0f) {
        int e;
        float mant = frexpf(v, &e);
        float mq   = rintf(mant * 1024.0f) * (1.0f / 1024.0f);
        ew = ldexpf(1.0f, e);
        b2 = ldexpf(mq + LMUL_OFFSET, e);
    }
    Bp[(size_t)m * KT + k]      = (_Float16)ew;
    Bp[(size_t)m * KT + TK + k] = (_Float16)b2;
}

// ---------------------------------------------------------------------------
// GEMM: out = A' * B'^T + bias  using v_wmma_f32_16x16x32_f16
// One wave -> 16x32 output tile (two accumulators sharing the A fragment).
// Grid sized exactly: 256 waves, no divergence -> EXEC all-ones for WMMA.
// ---------------------------------------------------------------------------
__global__ __launch_bounds__(256) void lmul_wmma_gemm(
    const _Float16* __restrict__ Ap,   // TN x KT
    const _Float16* __restrict__ Bp,   // TM x KT
    const float*    __restrict__ bias, // TM
    float*          __restrict__ out)  // TN x TM
{
    const int lane = threadIdx.x & 31;
    const int wid  = threadIdx.x >> 5;
    const int gw   = blockIdx.x * 8 + wid;  // 0..255
    const int tn   = gw >> 4;               // token tile   (16 tiles of 16 rows)
    const int tm   = gw & 15;               // feature tile (16 tiles of 32 cols)
    const int hf   = lane >> 4;             // lane half (0/1)
    const int lr   = lane & 15;

    // A fragment (16x32, row-major source): row M = lr, contiguous K chunk [16*hf, 16*hf+16)
    const _Float16* aRow = Ap + (size_t)(tn * 16 + lr) * KT;
    // B fragments (32x16, col N = lr, contiguous K chunk [16*hf, 16*hf+16))
    const int col0 = tm * 32 + lr;
    const int col1 = col0 + 16;
    const _Float16* b0Row = Bp + (size_t)col0 * KT + hf * 16;
    const _Float16* b1Row = Bp + (size_t)col1 * KT + hf * 16;

    float8 c0 = {};
    float8 c1 = {};

#pragma unroll 4
    for (int k = 0; k < KT; k += 32) {
        half16 a = *(const half16*)(aRow + k + hf * 16);
        half16 b0 = *(const half16*)(b0Row + k);
        half16 b1 = *(const half16*)(b1Row + k);

        c0 = __builtin_amdgcn_wmma_f32_16x16x32_f16(
                 false, a, false, b0, (short)0, c0, false, false);
        c1 = __builtin_amdgcn_wmma_f32_16x16x32_f16(
                 false, a, false, b1, (short)0, c1, false, false);
    }

    const float bb0 = bias[col0];
    const float bb1 = bias[col1];

    // C/D layout: VGPR g -> row (g + 8*hf), col lr  -> coalesced 64B runs
#pragma unroll
    for (int g = 0; g < 8; ++g) {
        const int row = tn * 16 + g + 8 * hf;
        out[(size_t)row * TM + col0] = c0[g] + bb0;
        out[(size_t)row * TM + col1] = c1[g] + bb1;
    }
}

// ---------------------------------------------------------------------------
extern "C" void kernel_launch(void* const* d_in, const int* in_sizes, int n_in,
                              void* d_out, int out_size, void* d_ws, size_t ws_size,
                              hipStream_t stream) {
    const float* x    = (const float*)d_in[0];   // TN*TK
    const float* wgt  = (const float*)d_in[1];   // TM*TK
    const float* bias = (const float*)d_in[2];   // TM
    float* out = (float*)d_out;                  // TN*TM

    // Workspace partition: A' (512 KB) then B' (1 MB)
    char* ws = (char*)d_ws;
    _Float16* Ap = (_Float16*)ws;
    _Float16* Bp = (_Float16*)(ws + (size_t)TN * KT * sizeof(_Float16));

    {
        int total = TN * TK;
        lmul_prep_x<<<(total + 255) / 256, 256, 0, stream>>>(x, Ap);
    }
    {
        int total = TM * TK;
        lmul_prep_w<<<(total + 255) / 256, 256, 0, stream>>>(wgt, Bp);
    }
    // 256 waves total: (TN/16) * (TM/32) = 16*16 tiles; 8 waves/block -> 32 blocks
    lmul_wmma_gemm<<<32, 256, 0, stream>>>(Ap, Bp, bias, out);
}
